// EnhancedMolecularGraph_61701500174913
// MI455X (gfx1250) — compile-verified, matrix operand order checked by reading, not run on hardware
//
#include <hip/hip_runtime.h>
#include <math.h>

typedef __attribute__((ext_vector_type(16))) _Float16 v16h;
typedef __attribute__((ext_vector_type(8)))  float    v8f;
typedef __attribute__((ext_vector_type(4)))  int      v4i;

#define HDIM 64
#define FN 32
#define FE 16
#define EPSBN 1e-5f
#define EPB 128            // edges per conv block (8 waves x 16)
#define BSLAB 4096         // f16 elements of B staged per i-iteration (8KB)

#if defined(__has_builtin)
#if __has_builtin(__builtin_amdgcn_global_load_async_to_lds_b128) && \
    __has_builtin(__builtin_amdgcn_s_wait_asynccnt)
#define HAVE_ASYNC_G2L 1
#endif
#endif

#if defined(HAVE_ASYNC_G2L)
#define ASYNC_WAIT(n) __builtin_amdgcn_s_wait_asynccnt(n)
typedef __attribute__((address_space(1))) v4i* gv4i_p;
typedef __attribute__((address_space(3))) v4i* lv4i_p;
#else
#define ASYNC_WAIT(n)
#endif

__device__ __forceinline__ void stage16(const void* g, void* l) {
#if defined(HAVE_ASYNC_G2L)
  __builtin_amdgcn_global_load_async_to_lds_b128(
      (gv4i_p)(unsigned long long)(size_t)g,
      (lv4i_p)(unsigned int)(size_t)l, 0, 0);
#else
  *(float4*)l = *(const float4*)g;
#endif
}

__device__ __forceinline__ v8f wmma32(v16h a, v16h b, v8f c) {
  return __builtin_amdgcn_wmma_f32_16x16x32_f16(false, a, false, b, (short)0, c, false, false);
}
// 16-bit A/B operand k index for lane-half `hi`, packed pair j (ISA 7.12.2)
__device__ __forceinline__ int kloc(int j, int hi) {
  int v = j >> 1, odd = j & 1;
  return 2 * v + ((v >= 4) ? 8 : 0) + 8 * hi + odd;
}
__device__ __forceinline__ float bnf(float x, float g, float b, float rm, float rv) {
  return (x - rm) * rsqrtf(rv + EPSBN) * g + b;
}
__device__ __forceinline__ float lrelu(float v) { return v > 0.f ? v : 0.2f * v; }
__device__ __forceinline__ void atomAdd(float* p, float v) {
  __hip_atomic_fetch_add(p, v, __ATOMIC_RELAXED, __HIP_MEMORY_SCOPE_AGENT);
}
__device__ __forceinline__ void atomMaxF(float* addr, float val) {
  if (val >= 0.f) atomicMax((int*)addr, __float_as_int(val));
  else            atomicMin((unsigned int*)addr, __float_as_uint(val));
}

__global__ void k_fill(float* p, float v, int n) {
  int i = blockIdx.x * blockDim.x + threadIdx.x;
  if (i < n) p[i] = v;
}

// ---------------- encoder: h = relu(bn(x@w1+b1)) @ w2 + b2 ; one wave / 16 nodes
__global__ void k_encoder(const float* __restrict__ x,
                          const float* __restrict__ w1, const float* __restrict__ b1,
                          const float* __restrict__ bg, const float* __restrict__ bb,
                          const float* __restrict__ brm, const float* __restrict__ brv,
                          const float* __restrict__ w2, const float* __restrict__ b2,
                          float* __restrict__ hout) {
  __shared__ float xs[16][FN];
  __shared__ _Float16 t1[16][HDIM];
  int lane = threadIdx.x, lo = lane & 15, hi = lane >> 4;
  int r0 = blockIdx.x * 16;
  for (int idx = lane; idx < 16 * FN; idx += 32) {
    int r = idx / FN, c = idx % FN;
    xs[r][c] = x[(size_t)(r0 + r) * FN + c];
  }
  __syncthreads();
  v16h A;
#pragma unroll
  for (int j = 0; j < 16; j++) A[j] = (_Float16)xs[lo][kloc(j, hi)];
#pragma unroll
  for (int nt = 0; nt < 4; nt++) {
    int n = nt * 16 + lo;
    v16h B;
#pragma unroll
    for (int j = 0; j < 16; j++) B[j] = (_Float16)w1[kloc(j, hi) * HDIM + n];
    v8f c = {};
    c = wmma32(A, B, c);
    float g = bg[n], b_ = bb[n], rm = brm[n], rv = brv[n], bias = b1[n];
#pragma unroll
    for (int v = 0; v < 8; v++) {
      int m = v + 8 * hi;
      float t = bnf(c[v] + bias, g, b_, rm, rv);
      t1[m][n] = (_Float16)fmaxf(t, 0.f);
    }
  }
  __syncthreads();
  v16h A0, A1;
#pragma unroll
  for (int j = 0; j < 16; j++) { A0[j] = t1[lo][kloc(j, hi)]; A1[j] = t1[lo][32 + kloc(j, hi)]; }
#pragma unroll
  for (int nt = 0; nt < 4; nt++) {
    int n = nt * 16 + lo;
    v16h B0, B1;
#pragma unroll
    for (int j = 0; j < 16; j++) {
      int k = kloc(j, hi);
      B0[j] = (_Float16)w2[k * HDIM + n];
      B1[j] = (_Float16)w2[(32 + k) * HDIM + n];
    }
    v8f c = {};
    c = wmma32(A0, B0, c);
    c = wmma32(A1, B1, c);
    float bias = b2[n];
#pragma unroll
    for (int v = 0; v < 8; v++)
      hout[(size_t)(r0 + v + 8 * hi) * HDIM + n] = c[v] + bias;
  }
}

// ---------------- edge hidden: eh = relu(bn(ea@w1+b1)); store in WMMA-A packed f16
__global__ void k_edgehidden(const float* __restrict__ ea,
                             const float* __restrict__ w1, const float* __restrict__ b1,
                             const float* __restrict__ bg, const float* __restrict__ bb,
                             const float* __restrict__ brm, const float* __restrict__ brv,
                             _Float16* __restrict__ ehpack) {
  __shared__ float es[16][FE];
  __shared__ _Float16 t1[16][HDIM];
  int lane = threadIdx.x, lo = lane & 15, hi = lane >> 4;
  int tile = blockIdx.x, r0 = tile * 16;
  for (int idx = lane; idx < 16 * FE; idx += 32) {
    int r = idx / FE, c = idx % FE;
    es[r][c] = ea[(size_t)(r0 + r) * FE + c];
  }
  __syncthreads();
  v16h A;
#pragma unroll
  for (int j = 0; j < 16; j++) { int k = kloc(j, hi); A[j] = (k < FE) ? (_Float16)es[lo][k] : (_Float16)0.f; }
#pragma unroll
  for (int nt = 0; nt < 4; nt++) {
    int n = nt * 16 + lo;
    v16h B;
#pragma unroll
    for (int j = 0; j < 16; j++) { int k = kloc(j, hi); B[j] = (k < FE) ? (_Float16)w1[k * HDIM + n] : (_Float16)0.f; }
    v8f c = {};
    c = wmma32(A, B, c);
    float g = bg[n], b_ = bb[n], rm = brm[n], rv = brv[n], bias = b1[n];
#pragma unroll
    for (int v = 0; v < 8; v++) {
      int m = v + 8 * hi;
      float t = bnf(c[v] + bias, g, b_, rm, rv);
      t1[m][n] = (_Float16)fmaxf(t, 0.f);
    }
  }
  __syncthreads();
  v16h* out = (v16h*)ehpack;
#pragma unroll
  for (int half = 0; half < 2; half++) {
    v16h p;
#pragma unroll
    for (int j = 0; j < 16; j++) p[j] = t1[lo][half * 32 + kloc(j, hi)];
    out[(size_t)(tile * 2 + half) * 32 + lane] = p;
  }
}

// ---------------- prepack edge_w2[64,4096] f32 -> f16 B-operand layout (one time)
__global__ void k_w2pack(const float* __restrict__ w2, _Float16* __restrict__ pack) {
  int idx = blockIdx.x * blockDim.x + threadIdx.x; // < 4096*64
  if (idx >= HDIM * HDIM * HDIM) return;
  int j = idx & 15, lane = (idx >> 4) & 31, half = (idx >> 9) & 1, t = idx >> 10;
  int lo = lane & 15, hi = lane >> 4;
  int k = half * 32 + kloc(j, hi);
  int c = t * 16 + lo;
  pack[idx] = (_Float16)w2[(size_t)k * (HDIM * HDIM) + c];
}

__global__ void k_deg(const int* __restrict__ ei, float* __restrict__ deg, int E) {
  int e = blockIdx.x * blockDim.x + threadIdx.x;
  if (e < E) atomAdd(&deg[ei[E + e]], 1.f);
}

// ---------------- fused NNConv message kernel.
// 8 waves / block, 16 edges / wave. Per i-iteration an 8KB slab of the
// prepacked edge_w2 is staged into LDS (async DMA when available, double
// buffered) and shared by all 8 waves; each wave runs 8 v_wmma_f32_16x16x32_f16
// per slab and folds the EW tile into msg = h_src @ EW with per-lane FMAs.
// Bias term (h_src @ B2) is precomputed node-wise (hb2) and added at scatter.
__global__ void __launch_bounds__(256) k_convmsg(
    const float* __restrict__ hin, const _Float16* __restrict__ ehpack,
    const _Float16* __restrict__ w2pack, const float* __restrict__ hb2,
    const int* __restrict__ ei, float* __restrict__ accum, int E) {
  __shared__ int ssrc[EPB], sdst[EPB];
  __shared__ float hs[EPB][HDIM];          // gathered source features (32KB)
  __shared__ _Float16 Bs[2][BSLAB];        // double-buffered B slab (16KB)
  int tid = threadIdx.x;
  int lane = tid & 31, wv = tid >> 5;
  int lo = lane & 15, hi = lane >> 4;
  int blk = blockIdx.x;
  int e0 = blk * EPB;
  if (tid < EPB) { ssrc[tid] = ei[e0 + tid]; sdst[tid] = ei[E + e0 + tid]; }
  // prologue: stage B slab for i=0 (overlaps with h gather below)
  {
    const char* g = (const char*)w2pack + (size_t)tid * 32;
    char* l = (char*)&Bs[0][0] + (size_t)tid * 32;
    stage16(g, l); stage16(g + 16, l + 16);
  }
  __syncthreads();
  for (int r = wv; r < EPB; r += 8)
    for (int c = lane; c < HDIM; c += 32) hs[r][c] = hin[(size_t)ssrc[r] * HDIM + c];
  // A operand (this wave's 16 edges)
  int etile = blk * 8 + wv;
  const v16h* Ap = (const v16h*)ehpack;
  v16h A0 = Ap[(size_t)(etile * 2 + 0) * 32 + lane];
  v16h A1 = Ap[(size_t)(etile * 2 + 1) * 32 + lane];
  int sbase = wv * 16;
  v8f acc[4] = {{}, {}, {}, {}};
  for (int i = 0; i < HDIM; i++) {
    int cb = i & 1;
    if (i + 1 < HDIM) {
      const char* g = (const char*)w2pack + (size_t)(i + 1) * (BSLAB * 2) + (size_t)tid * 32;
      char* l = (char*)&Bs[cb ^ 1][0] + (size_t)tid * 32;
#if !defined(HAVE_ASYNC_G2L)
      if (i + 2 < HDIM)
        __builtin_prefetch((const char*)w2pack + (size_t)(i + 2) * (BSLAB * 2) + (size_t)tid * 32, 0, 1);
#endif
      stage16(g, l); stage16(g + 16, l + 16);
      ASYNC_WAIT(2);       // current slab complete; next slab may be in flight
    } else {
      ASYNC_WAIT(0);
    }
    __syncthreads();       // slab + h gather visible to all waves
    float hv[8];
#pragma unroll
    for (int v = 0; v < 8; v++) hv[v] = hs[sbase + v + 8 * hi][i];
    const v16h* Bl = (const v16h*)&Bs[cb][0];
#pragma unroll
    for (int ot = 0; ot < 4; ot++) {
      v16h B0 = Bl[(ot * 2 + 0) * 32 + lane];
      v16h B1 = Bl[(ot * 2 + 1) * 32 + lane];
      v8f c = {};
      c = wmma32(A0, B0, c);
      c = wmma32(A1, B1, c);
#pragma unroll
      for (int v = 0; v < 8; v++) acc[ot][v] += hv[v] * c[v];
    }
    __syncthreads();       // all waves done reading Bs[cb] before restage
  }
#pragma unroll
  for (int ot = 0; ot < 4; ot++) {
#pragma unroll
    for (int v = 0; v < 8; v++) {
      int m = sbase + v + 8 * hi, o = ot * 16 + lo;
      float val = acc[ot][v] + hb2[(size_t)ssrc[m] * HDIM + o];
      atomAdd(&accum[(size_t)sdst[m] * HDIM + o], val);
    }
  }
}

__global__ void k_convfinish(const float* __restrict__ accum, const float* __restrict__ deg,
                             const float* __restrict__ hin, const float* __restrict__ root,
                             const float* __restrict__ bias,
                             const float* bg, const float* bb, const float* brm, const float* brv,
                             float* __restrict__ hout, int N) {
  int idx = blockIdx.x * blockDim.x + threadIdx.x;
  if (idx >= N * HDIM) return;
  int n = idx >> 6, o = idx & 63;
  float s = 0.f;
  for (int i = 0; i < HDIM; i++) s += hin[(size_t)n * HDIM + i] * root[i * HDIM + o];
  float v = accum[idx] / fmaxf(deg[n], 1.f) + s + bias[o];
  v = (v > 0.f) ? v : (expf(v) - 1.f);  // ELU
  hout[idx] = bnf(v, bg[o], bb[o], brm[o], brv[o]);
}

// ---------------- small node-level GEMM: out[n,o] = sum_i hin[n,i]*w[i*64+o]
__global__ void k_xh(const float* __restrict__ hin, const float* __restrict__ w,
                     float* __restrict__ xh, int N) {
  int idx = blockIdx.x * blockDim.x + threadIdx.x;
  if (idx >= N * HDIM) return;
  int n = idx >> 6, o = idx & 63;
  float s = 0.f;
  for (int i = 0; i < HDIM; i++) s += hin[(size_t)n * HDIM + i] * w[i * HDIM + o];
  xh[idx] = s;
}
__global__ void k_scores(const float* __restrict__ xh, const float* __restrict__ as,
                         const float* __restrict__ ad, float* __restrict__ a_src,
                         float* __restrict__ a_dst, int N) {
  int idx = blockIdx.x * blockDim.x + threadIdx.x;
  if (idx >= N * 8) return;
  int n = idx >> 3, hd = idx & 7;
  float s = 0.f, d = 0.f;
  for (int k = 0; k < 8; k++) {
    float v = xh[(size_t)n * HDIM + hd * 8 + k];
    s += v * as[hd * 8 + k];
    d += v * ad[hd * 8 + k];
  }
  a_src[idx] = s; a_dst[idx] = d;
}
__global__ void k_amax_init(const float* a_src, const float* a_dst, float* amax, int N) {
  int idx = blockIdx.x * blockDim.x + threadIdx.x;
  if (idx < N * 8) amax[idx] = lrelu(a_src[idx] + a_dst[idx]);  // self loop
}
__global__ void k_edge_amax(const int* __restrict__ ei, const float* a_src, const float* a_dst,
                            float* amax, int E) {
  int idx = blockIdx.x * blockDim.x + threadIdx.x;
  if (idx >= E * 8) return;
  int e = idx >> 3, hd = idx & 7;
  int s = ei[e], d = ei[E + e];
  atomMaxF(&amax[(size_t)d * 8 + hd], lrelu(a_src[(size_t)s * 8 + hd] + a_dst[(size_t)d * 8 + hd]));
}
__global__ void k_asum(const int* __restrict__ ei, const float* a_src, const float* a_dst,
                       const float* amax, float* asum, int E, int N) {
  int idx = blockIdx.x * blockDim.x + threadIdx.x;
  if (idx >= (E + N) * 8) return;
  int ep = idx >> 3, hd = idx & 7, s, d;
  if (ep < E) { s = ei[ep]; d = ei[E + ep]; } else { s = d = ep - E; }
  float v = lrelu(a_src[(size_t)s * 8 + hd] + a_dst[(size_t)d * 8 + hd]);
  atomAdd(&asum[(size_t)d * 8 + hd], expf(v - amax[(size_t)d * 8 + hd]));
}
__global__ void k_hg_init(const float* bias, float* hg, int N) {
  int idx = blockIdx.x * blockDim.x + threadIdx.x;
  if (idx < N * HDIM) hg[idx] = bias[idx & 63];
}
__global__ void k_gat_accum(const int* __restrict__ ei, const float* __restrict__ xh,
                            const float* a_src, const float* a_dst, const float* amax,
                            const float* asum, float* __restrict__ hg, int E, int N) {
  int idx = blockIdx.x * blockDim.x + threadIdx.x;
  if (idx >= (E + N) * 8) return;
  int ep = idx >> 3, hd = idx & 7, s, d;
  if (ep < E) { s = ei[ep]; d = ei[E + ep]; } else { s = d = ep - E; }
  float v = lrelu(a_src[(size_t)s * 8 + hd] + a_dst[(size_t)d * 8 + hd]);
  float alpha = expf(v - amax[(size_t)d * 8 + hd]) / asum[(size_t)d * 8 + hd];
  for (int k = 0; k < 8; k++)
    atomAdd(&hg[(size_t)d * HDIM + hd * 8 + k], alpha * xh[(size_t)s * HDIM + hd * 8 + k]);
}

// ---------------- readout ----------------
__global__ void k_readout(const float* __restrict__ h, const int* __restrict__ batch,
                          float* gsum, float* gcnt, int N) {
  int idx = blockIdx.x * blockDim.x + threadIdx.x;
  if (idx >= N * HDIM) return;
  int n = idx >> 6, o = idx & 63;
  int g = batch[n];
  atomAdd(&gsum[(size_t)g * HDIM + o], h[idx]);
  if (o == 0) atomAdd(&gcnt[g], 1.f);
}
__global__ void k_mlp(const float* __restrict__ gsum, const float* __restrict__ gcnt,
                      const float* w1, const float* b1,
                      const float* bg, const float* bb, const float* brm, const float* brv,
                      const float* w2, const float* b2, const float* w3, const float* b3,
                      float* __restrict__ out) {
  __shared__ float gf[2 * HDIM];
  __shared__ float o1[HDIM];
  __shared__ float o2[HDIM / 2];
  int g = blockIdx.x, t = threadIdx.x;
  float cnt = fmaxf(gcnt[g], 1.f);
  float sv = gsum[(size_t)g * HDIM + t];
  gf[t] = sv / cnt; gf[HDIM + t] = sv;
  __syncthreads();
  float s = 0.f;
  for (int i = 0; i < 2 * HDIM; i++) s += gf[i] * w1[i * HDIM + t];
  o1[t] = fmaxf(bnf(s + b1[t], bg[t], bb[t], brm[t], brv[t]), 0.f);
  __syncthreads();
  if (t < 32) {
    float s2 = 0.f;
    for (int i = 0; i < HDIM; i++) s2 += o1[i] * w2[i * 32 + t];
    o2[t] = fmaxf(s2 + b2[t], 0.f);
  }
  __syncthreads();
  if (t == 0) {
    float s3 = 0.f;
    for (int i = 0; i < 32; i++) s3 += o2[i] * w3[i];
    out[g] = s3 + b3[0];
  }
}

extern "C" void kernel_launch(void* const* d_in, const int* in_sizes, int n_in,
                              void* d_out, int out_size, void* d_ws, size_t ws_size,
                              hipStream_t stream) {
  const float* x     = (const float*)d_in[0];
  const int*   ei    = (const int*)d_in[1];
  const float* ea    = (const float*)d_in[2];
  const int*   batch = (const int*)d_in[3];
  const float* p[50];
  for (int i = 4; i < 50; i++) p[i] = (const float*)d_in[i];

  const int N = in_sizes[0] / FN;
  const int E = in_sizes[2] / FE;
  const int G = out_size;

  // workspace carve
  size_t off = 0;
  auto carve = [&](size_t bytes) {
    void* ptr = (char*)d_ws + off;
    off = (off + bytes + 255) & ~(size_t)255;
    return ptr;
  };
  float*    bufA   = (float*)carve((size_t)N * HDIM * 4);   // h0 / h2
  float*    bufB   = (float*)carve((size_t)N * HDIM * 4);   // h1 / hg2
  float*    bufC   = (float*)carve((size_t)N * HDIM * 4);   // hg1
  float*    accum  = (float*)carve((size_t)N * HDIM * 4);
  float*    xh     = (float*)carve((size_t)N * HDIM * 4);   // reused: hb2 / GAT xh
  _Float16* ehpack = (_Float16*)carve((size_t)E * HDIM * 2);
  _Float16* w2pack = (_Float16*)carve((size_t)HDIM * HDIM * HDIM * 2);
  float*    deg    = (float*)carve((size_t)N * 4);
  float*    a_src  = (float*)carve((size_t)N * 8 * 4);
  float*    a_dst  = (float*)carve((size_t)N * 8 * 4);
  float*    amax   = (float*)carve((size_t)N * 8 * 4);
  float*    asum   = (float*)carve((size_t)N * 8 * 4);
  float*    gsum   = (float*)carve((size_t)G * HDIM * 4);
  float*    gcnt   = (float*)carve((size_t)G * 4);
  (void)ws_size; (void)n_in;

  auto cdiv = [](int a, int b) { return (a + b - 1) / b; };

  // encoder + edge network prep
  k_encoder<<<N / 16, 32, 0, stream>>>(x, p[4], p[5], p[6], p[7], p[8], p[9], p[10], p[11], bufA);
  k_edgehidden<<<E / 16, 32, 0, stream>>>(ea, p[12], p[13], p[14], p[15], p[16], p[17], ehpack);
  k_w2pack<<<cdiv(HDIM * HDIM * HDIM, 256), 256, 0, stream>>>(p[18], w2pack);
  k_fill<<<cdiv(N, 256), 256, 0, stream>>>(deg, 0.f, N);
  k_deg<<<cdiv(E, 256), 256, 0, stream>>>(ei, deg, E);

  auto conv = [&](const float* hin, const float* root, const float* bias,
                  const float* g_, const float* b_, const float* rm_, const float* rv_,
                  float* hout) {
    k_fill<<<cdiv(N * HDIM, 256), 256, 0, stream>>>(accum, 0.f, N * HDIM);
    k_xh<<<cdiv(N * HDIM, 256), 256, 0, stream>>>(hin, p[19], xh, N);  // hb2 = hin @ B2
    k_convmsg<<<E / EPB, 256, 0, stream>>>(hin, ehpack, w2pack, xh, ei, accum, E);
    k_convfinish<<<cdiv(N * HDIM, 256), 256, 0, stream>>>(accum, deg, hin, root, bias,
                                                          g_, b_, rm_, rv_, hout, N);
  };
  auto gat = [&](const float* hin, const float* w, const float* as, const float* ad,
                 const float* bias, float* hout) {
    k_xh<<<cdiv(N * HDIM, 256), 256, 0, stream>>>(hin, w, xh, N);
    k_scores<<<cdiv(N * 8, 256), 256, 0, stream>>>(xh, as, ad, a_src, a_dst, N);
    k_amax_init<<<cdiv(N * 8, 256), 256, 0, stream>>>(a_src, a_dst, amax, N);
    k_edge_amax<<<cdiv(E * 8, 256), 256, 0, stream>>>(ei, a_src, a_dst, amax, E);
    k_fill<<<cdiv(N * 8, 256), 256, 0, stream>>>(asum, 0.f, N * 8);
    k_asum<<<cdiv((E + N) * 8, 256), 256, 0, stream>>>(ei, a_src, a_dst, amax, asum, E, N);
    k_hg_init<<<cdiv(N * HDIM, 256), 256, 0, stream>>>(bias, hout, N);
    k_gat_accum<<<cdiv((E + N) * 8, 256), 256, 0, stream>>>(ei, xh, a_src, a_dst, amax, asum,
                                                            hout, E, N);
  };

  conv(bufA, p[20], p[21], p[24], p[25], p[26], p[27], bufB);            // h1 (elu+bn1)
  gat(bufB, p[32], p[33], p[34], p[35], bufC);                           // gat1 -> hg1
  conv(bufC, p[22], p[23], p[28], p[29], p[30], p[31], bufA);            // h2 (elu+bn2)
  gat(bufA, p[36], p[37], p[38], p[39], bufB);                           // gat2 -> hg2

  k_fill<<<cdiv(G * HDIM, 256), 256, 0, stream>>>(gsum, 0.f, G * HDIM);
  k_fill<<<cdiv(G, 256), 256, 0, stream>>>(gcnt, 0.f, G);
  k_readout<<<cdiv(N * HDIM, 256), 256, 0, stream>>>(bufB, batch, gsum, gcnt, N);
  k_mlp<<<G, 64, 0, stream>>>(gsum, gcnt, p[40], p[41], p[42], p[43], p[44], p[45],
                              p[46], p[47], p[48], p[49], (float*)d_out);
}
